// Encoder_8564164788526
// MI455X (gfx1250) — matrix operand order repaired and checked by reading, not compiled, not run on hardware
//
#include <hip/hip_runtime.h>
#include <hip/hip_bf16.h>
#include <math.h>

typedef __bf16 bf16;
typedef __attribute__((ext_vector_type(16))) __bf16 v16bf;
typedef __attribute__((ext_vector_type(8)))  __bf16 v8bf;
typedef __attribute__((ext_vector_type(8)))  float  v8f;
typedef int v4i __attribute__((vector_size(16)));

#define BATCH 64
#define SEQ   512
#define HID   512
#define EMBD  300
#define G3    1536            // 3*HID
#define MTOT  (BATCH*SEQ)     // 32768

// LDS staging geometry for the big GEMM (operand-order layout):
//   per 16x32 bf16 tile: 32 operand-lanes * 32B, padded to 48B/lane stride
#define TILE_B 1536           // 32 * 48
#define ABYTES 12288          // 8 tiles (128 rows x 32 K)
#define BUFB   24576          // A panel + B panel
// double buffered -> 48 KB LDS per workgroup (WGP has 320 KB)

typedef __attribute__((address_space(1))) v4i gv4i;   // global int4
typedef __attribute__((address_space(3))) v4i lv4i;   // LDS int4

// ---------------------------------------------------------------------------
// CDNA5 async global->LDS copy of one 16B chunk (per-lane dst), with fallback.
__device__ __forceinline__ void async_copy16(const bf16* g, bf16* l) {
#if __has_builtin(__builtin_amdgcn_global_load_async_to_lds_b128)
  __builtin_amdgcn_global_load_async_to_lds_b128((gv4i*)g, (lv4i*)l, 0, 0);
#else
  *(v8bf*)l = *(const v8bf*)g;   // sync fallback (DScnt handled by barrier)
#endif
}

template <int N> __device__ __forceinline__ void wait_async() {
#if __has_builtin(__builtin_amdgcn_s_wait_asynccnt)
  __builtin_amdgcn_s_wait_asynccnt(N);
#else
  asm volatile("s_wait_asynccnt %0" ::"i"(N) : "memory");
#endif
}

// ---------------------------------------------------------------------------
// WMMA operand loaders (bf16 16x16x32, wave32).
//
// A (16x32 MxK) row-major global, leading dim lda:
//   lanes 0-15  : row M=lane,    K [0..15]
//   lanes 16-31 : row M=lane-16, K [16..31]
__device__ __forceinline__ v16bf load_a16x32(const bf16* A, int lda, int row0,
                                             int k0, int lane) {
  int m  = row0 + (lane & 15);
  int kh = (lane >> 4) & 1;
  const bf16* p = A + (size_t)m * lda + k0 + kh * 16;
  v8bf lo = *(const v8bf*)(p);
  v8bf hi = *(const v8bf*)(p + 8);
  return __builtin_shufflevector(lo, hi, 0, 1, 2, 3, 4, 5, 6, 7,
                                 8, 9, 10, 11, 12, 13, 14, 15);
}

// B (32x16 KxN), B[k][n] = W[n][k] with W row-major [N,K] leading dim ldw:
//   lane n in 0-15 : K=[0..15] of W row n;  lane n+16 : K=[16..31]
__device__ __forceinline__ v16bf load_b32x16(const bf16* W, int ldw, int col0,
                                             int k0, int lane) {
  int n  = col0 + (lane & 15);
  int kh = (lane >> 4) & 1;
  const bf16* p = W + (size_t)n * ldw + k0 + kh * 16;
  return *(const v16bf*)p;
}

// Read one staged operand (32B at lane*48) out of LDS: two ds_load_b128.
__device__ __forceinline__ v16bf lds_read_op(const char* base, int tile,
                                             int lane) {
  const char* p = base + tile * TILE_B + lane * 48;
  v8bf lo = *(const v8bf*)p;
  v8bf hi = *(const v8bf*)(p + 16);
  return __builtin_shufflevector(lo, hi, 0, 1, 2, 3, 4, 5, 6, 7,
                                 8, 9, 10, 11, 12, 13, 14, 15);
}

// Stage a 128x32 A panel + 128x32 B panel into LDS in operand order.
// 1024 16B chunks / 256 threads = 4 async instructions per thread.
__device__ __forceinline__ void stage_panels(const bf16* __restrict__ A, int lda,
                                             int rowBase,
                                             const bf16* __restrict__ W, int ldw,
                                             int colBase, int k0, char* ldsBuf,
                                             int tid) {
#pragma unroll
  for (int i = 0; i < 4; ++i) {
    int c    = i * 256 + tid;        // 0..1023
    int isB  = c >> 9;
    int cc   = c & 511;
    int tile = cc >> 6;              // 0..7
    int ln   = (cc >> 1) & 31;       // operand lane
    int h    = cc & 1;               // which 16B half of the lane's 32B
    int kh   = ln >> 4;
    int mn   = (isB ? colBase : rowBase) + tile * 16 + (ln & 15);
    const bf16* gsrc;
    if (isB) gsrc = W + (size_t)mn * ldw + k0 + kh * 16 + h * 8;
    else     gsrc = A + (size_t)mn * lda + k0 + kh * 16 + h * 8;
    char* ldst = ldsBuf + isB * ABYTES + tile * TILE_B + ln * 48 + h * 16;
    async_copy16(gsrc, (bf16*)ldst);
  }
}

// ---------------------------------------------------------------------------
// Weight conversion: f32 [rows,K] -> bf16 [rows,Kp] zero-padded.
__global__ void cvt_w_kernel(const float* __restrict__ src, bf16* __restrict__ dst,
                             int K, int Kp, int total) {
  int idx = blockIdx.x * blockDim.x + threadIdx.x;
  if (idx >= total) return;
  int r = idx / Kp, c = idx - r * Kp;
  float v = (c < K) ? src[(size_t)r * K + c] : 0.0f;
  dst[idx] = (bf16)v;
}

__global__ void embed_gather_kernel(const int* __restrict__ ids,
                                    const float* __restrict__ emb,
                                    bf16* __restrict__ x, int kp, int total) {
  int idx = blockIdx.x * blockDim.x + threadIdx.x;
  if (idx >= total) return;
  int row = idx / kp, c = idx - row * kp;
  float v = (c < EMBD) ? emb[(size_t)ids[row] * EMBD + c] : 0.0f;
  x[(size_t)row * HID + c] = (bf16)v;
}

__global__ void load_h0_kernel(const float* __restrict__ src,
                               float* __restrict__ hf, bf16* __restrict__ hb,
                               int n) {
  int i = blockIdx.x * blockDim.x + threadIdx.x;
  if (i >= n) return;
  float v = src[i];
  hf[i] = v;
  hb[i] = (bf16)v;
}

__global__ void copy_f32_kernel(const float* __restrict__ s,
                                float* __restrict__ d, int n) {
  int i = blockIdx.x * blockDim.x + threadIdx.x;
  if (i < n) d[i] = s[i];
}

// ---------------------------------------------------------------------------
// Big input-gate GEMM: C[M,G3] = A_bf16[M,Kloop] x W_bf16[G3,Kloop]^T + bias
// Block = 256 threads (8 waves) -> 128x128 tile; wave -> 32x64 (2x4 tiles).
// A/B panels double-buffered in LDS via async global->LDS loads.
__global__ __launch_bounds__(256) void wmma_gemm_bias_kernel(
    const bf16* __restrict__ A, int lda, const bf16* __restrict__ W, int ldw,
    const float* __restrict__ bias, float* __restrict__ C, int N, int Kloop) {
  __shared__ alignas(128) char lds[2 * BUFB];

  int tid  = threadIdx.x;
  int lane = tid & 31;
  int wid  = tid >> 5;               // 0..7
  int mg   = wid & 3;                // 4 m-groups x 32 rows
  int ng   = wid >> 2;               // 2 n-groups x 64 cols
  int rowBase = blockIdx.x * 128;
  int colBase = blockIdx.y * 128;
  int row0 = rowBase + mg * 32;
  int col0 = colBase + ng * 64;

  v8f acc[2][4] = {};
  int kIters = Kloop >> 5;

  stage_panels(A, lda, rowBase, W, ldw, colBase, 0, lds, tid);

  for (int it = 0; it < kIters; ++it) {
    char* cur = lds + (it & 1) * BUFB;
    if (it + 1 < kIters) {
      stage_panels(A, lda, rowBase, W, ldw, colBase, (it + 1) << 5,
                   lds + ((it + 1) & 1) * BUFB, tid);
      wait_async<4>();               // current panel done; next 4 in flight
    } else {
      wait_async<0>();
    }
    __syncthreads();

    v16bf a0 = lds_read_op(cur, mg * 2 + 0, lane);
    v16bf a1 = lds_read_op(cur, mg * 2 + 1, lane);
    const char* bbase = cur + ABYTES;
#pragma unroll
    for (int j = 0; j < 4; ++j) {
      v16bf b = lds_read_op(bbase, ng * 4 + j, lane);
      acc[0][j] = __builtin_amdgcn_wmma_f32_16x16x32_bf16(
          false, a0, false, b, (short)0, acc[0][j], false, false);
      acc[1][j] = __builtin_amdgcn_wmma_f32_16x16x32_bf16(
          false, a1, false, b, (short)0, acc[1][j], false, false);
    }
    __syncthreads();                 // protect buffer about to be restaged
  }

#pragma unroll
  for (int i = 0; i < 2; i++)
#pragma unroll
    for (int j = 0; j < 4; j++) {
      int col = col0 + j * 16 + (lane & 15);
      float bv = bias[col];
#pragma unroll
      for (int r = 0; r < 8; r++) {
        int row = row0 + i * 16 + r + ((lane >> 4) << 3);
        C[(size_t)row * N + col] = acc[i][j][r] + bv;
      }
    }
}

// ---------------------------------------------------------------------------
// Recurrent per-step GEMM: hg[64,G3] = h_bf16[64,HID] x Whh_bf16[G3,HID]^T + bhh
// 12 blocks x 8 waves = 96 waves; wave -> 16x64 tile. Whh is L2-resident.
__global__ __launch_bounds__(256) void wmma_step_hg_kernel(
    const bf16* __restrict__ hbf, const bf16* __restrict__ Whh,
    const float* __restrict__ bhh, float* __restrict__ hg) {
  int lane = threadIdx.x & 31;
  int w    = blockIdx.x * 8 + (threadIdx.x >> 5);    // 0..95
  int row0 = (w & 3) * 16;
  int col0 = (w >> 2) * 64;

  v8f acc[4] = {};

  for (int k0 = 0; k0 < HID; k0 += 32) {
    v16bf a = load_a16x32(hbf, HID, row0, k0, lane);
    __builtin_prefetch(Whh + (size_t)(col0 + (lane & 15)) * HID + k0 + 128, 0, 1);
#pragma unroll
    for (int j = 0; j < 4; j++) {
      v16bf b = load_b32x16(Whh, HID, col0 + j * 16, k0, lane);
      acc[j] = __builtin_amdgcn_wmma_f32_16x16x32_bf16(
          false, a, false, b, (short)0, acc[j], false, false);
    }
  }

#pragma unroll
  for (int j = 0; j < 4; j++) {
    int col = col0 + j * 16 + (lane & 15);
    float bv = bhh[col];
#pragma unroll
    for (int r = 0; r < 8; r++) {
      int row = row0 + r + ((lane >> 4) << 3);
      hg[(size_t)row * G3 + col] = acc[j][r] + bv;
    }
  }
}

// ---------------------------------------------------------------------------
// Fused GRU gates for one timestep over [B=64, H=512].
__global__ void gru_gate_kernel(const float* __restrict__ xg,
                                const float* __restrict__ hg,
                                const float* __restrict__ h_prev,
                                float* __restrict__ h_new,
                                bf16* __restrict__ h_new_bf,
                                bf16* __restrict__ out_bf,
                                float* __restrict__ out_f32, int t,
                                int write_f32) {
  int idx = blockIdx.x * blockDim.x + threadIdx.x;   // 0..32767
  int b = idx >> 9;
  int j = idx & 511;
  size_t xrow = ((size_t)b * SEQ + t) * G3;
  float xr = xg[xrow + j];
  float xz = xg[xrow + HID + j];
  float xn = xg[xrow + 2 * HID + j];
  size_t hrow = (size_t)b * G3;
  float hr = hg[hrow + j];
  float hz = hg[hrow + HID + j];
  float hn = hg[hrow + 2 * HID + j];
  float hp = h_prev[idx];
  float r = 1.0f / (1.0f + __expf(-(xr + hr)));
  float z = 1.0f / (1.0f + __expf(-(xz + hz)));
  float n = tanhf(xn + r * hn);
  float h = (1.0f - z) * n + z * hp;
  h_new[idx]    = h;
  h_new_bf[idx] = (bf16)h;
  size_t orow = ((size_t)b * SEQ + t) * HID + j;
  out_bf[orow] = (bf16)h;          // next layer's bf16 input sequence
  if (write_f32) out_f32[orow] = h;
}

// ---------------------------------------------------------------------------
extern "C" void kernel_launch(void* const* d_in, const int* in_sizes, int n_in,
                              void* d_out, int out_size, void* d_ws,
                              size_t ws_size, hipStream_t stream) {
  (void)in_sizes; (void)n_in; (void)out_size; (void)ws_size;

  const int*   ids    = (const int*)d_in[0];
  const float* hidden = (const float*)d_in[1];
  const float* emb    = (const float*)d_in[2];
  const float* Wih[3] = {(const float*)d_in[3], (const float*)d_in[7],  (const float*)d_in[11]};
  const float* Whh[3] = {(const float*)d_in[4], (const float*)d_in[8],  (const float*)d_in[12]};
  const float* bih[3] = {(const float*)d_in[5], (const float*)d_in[9],  (const float*)d_in[13]};
  const float* bhh[3] = {(const float*)d_in[6], (const float*)d_in[10], (const float*)d_in[14]};
  float* out = (float*)d_out;

  // ---- workspace carve-up --------------------------------------------------
  char* ws = (char*)d_ws;
  size_t off = 0;
  auto alloc = [&](size_t bytes) -> void* {
    void* p = ws + off;
    off = (off + bytes + 255) & ~(size_t)255;
    return p;
  };
  const int Kin[3] = {EMBD, HID, HID};
  const int Kp[3]  = {320, 512, 512};   // K padded to multiple of 32

  bf16* wih_b[3]; bf16* whh_b[3];
  for (int l = 0; l < 3; l++) {
    wih_b[l] = (bf16*)alloc((size_t)G3 * Kp[l] * sizeof(bf16));
    whh_b[l] = (bf16*)alloc((size_t)G3 * HID * sizeof(bf16));
  }
  bf16*  seq = (bf16*)alloc((size_t)MTOT * HID * sizeof(bf16));     // 33.5 MB
  float* xg  = (float*)alloc((size_t)MTOT * G3 * sizeof(float));    // 201 MB
  float* hg  = (float*)alloc((size_t)BATCH * G3 * sizeof(float));
  float* hf[2]; bf16* hb[2];
  for (int i = 0; i < 2; i++) {
    hf[i] = (float*)alloc((size_t)BATCH * HID * sizeof(float));
    hb[i] = (bf16*)alloc((size_t)BATCH * HID * sizeof(bf16));
  }

  // ---- one-time conversions ------------------------------------------------
  for (int l = 0; l < 3; l++) {
    int n1 = G3 * Kp[l];
    cvt_w_kernel<<<(n1 + 255) / 256, 256, 0, stream>>>(Wih[l], wih_b[l],
                                                       Kin[l], Kp[l], n1);
    int n2 = G3 * HID;
    cvt_w_kernel<<<(n2 + 255) / 256, 256, 0, stream>>>(Whh[l], whh_b[l],
                                                       HID, HID, n2);
  }
  {
    int n = MTOT * Kp[0];
    embed_gather_kernel<<<(n + 255) / 256, 256, 0, stream>>>(ids, emb, seq,
                                                             Kp[0], n);
  }

  // ---- layers --------------------------------------------------------------
  for (int l = 0; l < 3; l++) {
    // xg = seq @ Wih^T + bih for all timesteps (WMMA GEMM, async-LDS staged)
    dim3 g(MTOT / 128, G3 / 128);
    wmma_gemm_bias_kernel<<<g, 256, 0, stream>>>(seq, HID, wih_b[l], Kp[l],
                                                 bih[l], xg, G3, Kp[l]);

    load_h0_kernel<<<(BATCH * HID + 255) / 256, 256, 0, stream>>>(
        hidden + (size_t)l * BATCH * HID, hf[0], hb[0], BATCH * HID);

    int cur = 0;
    for (int t = 0; t < SEQ; t++) {
      wmma_step_hg_kernel<<<12, 256, 0, stream>>>(hb[cur], whh_b[l], bhh[l], hg);
      gru_gate_kernel<<<MTOT / 256, 256, 0, stream>>>(
          xg, hg, hf[cur], hf[1 - cur], hb[1 - cur], seq, out, t,
          (l == 2) ? 1 : 0);
      cur ^= 1;
    }
    copy_f32_kernel<<<(BATCH * HID + 255) / 256, 256, 0, stream>>>(
        hf[cur], out + (size_t)BATCH * SEQ * HID + (size_t)l * BATCH * HID,
        BATCH * HID);
  }
}
